// NAFBlock_16277926052067
// MI455X (gfx1250) — matrix operand order repaired and checked by reading, not compiled, not run on hardware
//
#include <hip/hip_runtime.h>
#include <hip/hip_bf16.h>

typedef __attribute__((ext_vector_type(16))) _Float16 v16h;
typedef __attribute__((ext_vector_type(8)))  _Float16 v8h;
typedef __attribute__((ext_vector_type(8)))  float    v8f;
typedef __attribute__((ext_vector_type(4)))  float    v4f;

#define EPS 1e-6f
#define NPIX (8*192*192)          // 294912
#define NTILE (NPIX/16)           // 18432 (16-pixel tiles)
#define WAVES_PER_BLK 4
#define NBLK (NTILE/WAVES_PER_BLK) // 4608
#define TILES_PER_IMG 2304
#define BLKS_PER_IMG (TILES_PER_IMG/WAVES_PER_BLK) // 576

// ---------------- workspace layout (bytes) ----------------
// gapBlk : f32[NBLK*128]      @ 0          (2359296 B)  per-block channel sums
// b1p    : f32[256]           @ 2359296    (w1 bias, LN1/dw folded)
// bf1p   : f32[256]           @ 2360320    (wf1 bias, LN2 folded)
// b2p    : f32[128]           @ 2361344    (beta*b2)
// bf2p   : f32[128]           @ 2361856    (gamma*bf2)
// w1p    : f16[32768]         @ 2362368    (B-swizzled, ln1s*wdw folded)
// wf1p   : f16[32768]         @ 2427904    (B-swizzled, ln2s folded)
// wf2p   : f16[16384]         @ 2493440    (B-swizzled, gamma folded)
// w2bp   : f16[8*16384]       @ 2526208    (B-swizzled, att*beta folded, per image)
// total ~2.79 MB

__device__ __forceinline__ float reduce16(float v) {
    v += __shfl_xor(v, 1); v += __shfl_xor(v, 2);
    v += __shfl_xor(v, 4); v += __shfl_xor(v, 8);
    return v;
}

// Build f16 A-operand (4 chunks of 16x32) from global fp32 rows with LayerNorm.
// A swizzle (ISA 7.12.2, 16-bit A 16x32): lane<16 holds row p, K blocks {0,2,4,...};
// lane+16 holds row p, K blocks {1,3,5,...}; per operand c: slots 0-7 = K 32c..32c+7,
// slots 8-15 = K 32c+16..32c+23 (lanes>=16 shifted by 8).
template <bool STASH>
__device__ __forceinline__ void build_A_global(const float* __restrict__ xp,
                                               float* xt, v16h A[4], int lane) {
    const int p = lane & 15, par = lane >> 4;
    const float* base = xp + p * 128 + par * 8;
    v4f arr[16];
    float s = 0.f, q = 0.f;
#pragma unroll
    for (int j = 0; j < 8; ++j) {
        v4f a0 = *(const v4f*)(base + j * 16);
        v4f a1 = *(const v4f*)(base + j * 16 + 4);
        arr[2 * j] = a0; arr[2 * j + 1] = a1;
        if (STASH) {
            int cs = j * 16 + par * 8;
            *(v4f*)(xt + p * 128 + cs) = a0;
            *(v4f*)(xt + p * 128 + cs + 4) = a1;
        }
    }
#pragma unroll
    for (int m = 0; m < 16; ++m) {
        s += arr[m][0] + arr[m][1] + arr[m][2] + arr[m][3];
        q += arr[m][0]*arr[m][0] + arr[m][1]*arr[m][1] + arr[m][2]*arr[m][2] + arr[m][3]*arr[m][3];
    }
    s += __shfl_xor(s, 16); q += __shfl_xor(q, 16);
    const float mu = s * (1.f / 128.f);
    const float rs = rsqrtf(q * (1.f / 128.f) - mu * mu + EPS);
#pragma unroll
    for (int c = 0; c < 4; ++c)
#pragma unroll
        for (int i = 0; i < 16; ++i)
            A[c][i] = (_Float16)((arr[4 * c + (i >> 2)][i & 3] - mu) * rs);
}

// Build A-operand from an LDS f16 [16][128] tile (post-gate activations).
__device__ __forceinline__ void build_A_ldsh(const _Float16* gt, v16h A[4], int lane) {
    const int p = lane & 15, par = lane >> 4;
#pragma unroll
    for (int c = 0; c < 4; ++c) {
        v8h lo = *(const v8h*)(gt + p * 128 + (4 * c + par) * 8);
        v8h hi = *(const v8h*)(gt + p * 128 + (4 * c + 2 + par) * 8);
#pragma unroll
        for (int i = 0; i < 8; ++i) { A[c][i] = lo[i]; A[c][8 + i] = hi[i]; }
    }
}

// Build A-operand from LDS fp32 [16][128] tile with LayerNorm (mu/rs per pixel).
__device__ __forceinline__ void build_A_ldsf(const float* xt, float mu, float rs,
                                             v16h A[4], int lane) {
    const int p = lane & 15, par = lane >> 4;
#pragma unroll
    for (int c = 0; c < 4; ++c) {
        const int c0 = 32 * c + 8 * par;
        v4f a = *(const v4f*)(xt + p * 128 + c0);
        v4f b = *(const v4f*)(xt + p * 128 + c0 + 4);
        v4f cq = *(const v4f*)(xt + p * 128 + c0 + 16);
        v4f d = *(const v4f*)(xt + p * 128 + c0 + 20);
#pragma unroll
        for (int i = 0; i < 4; ++i) {
            A[c][i]      = (_Float16)((a[i]  - mu) * rs);
            A[c][4 + i]  = (_Float16)((b[i]  - mu) * rs);
            A[c][8 + i]  = (_Float16)((cq[i] - mu) * rs);
            A[c][12 + i] = (_Float16)((d[i]  - mu) * rs);
        }
    }
}

// GEMM 16x128 @ 128x256 with SimpleGate: emit(nt, ch, g[8]) per 16x16 gated tile.
template <typename F>
__device__ __forceinline__ void gemm_gate(const v16h A[4], const _Float16* __restrict__ Wp,
                                          const float* __restrict__ bp, int lane, F&& emit) {
    const int ch0 = lane & 15;
#pragma unroll
    for (int nt = 0; nt < 8; ++nt) {
        v8f a1 = {}, a2 = {};
#pragma unroll
        for (int kc = 0; kc < 4; ++kc) {
            v16h b1 = *(const v16h*)(Wp + ((size_t)(kc * 16 + nt) * 32 + lane) * 16);
            v16h b2 = *(const v16h*)(Wp + ((size_t)(kc * 16 + nt + 8) * 32 + lane) * 16);
            a1 = __builtin_amdgcn_wmma_f32_16x16x32_f16(false, A[kc], false, b1, (short)0, a1, false, false);
            a2 = __builtin_amdgcn_wmma_f32_16x16x32_f16(false, A[kc], false, b2, (short)0, a2, false, false);
        }
        const int ch = nt * 16 + ch0;
        const float bb1 = bp[ch], bb2 = bp[128 + ch];
        float g[8];
#pragma unroll
        for (int r = 0; r < 8; ++r) g[r] = (a1[r] + bb1) * (a2[r] + bb2);
        emit(nt, ch, g);
    }
}

// GEMM 16x128 @ 128x128: emit(nt, ch, acc) per 16x16 fp32 tile.
template <typename F>
__device__ __forceinline__ void gemm8(const v16h A[4], const _Float16* __restrict__ Wp,
                                      int lane, F&& emit) {
    const int ch0 = lane & 15;
#pragma unroll
    for (int nt = 0; nt < 8; ++nt) {
        v8f acc = {};
#pragma unroll
        for (int kc = 0; kc < 4; ++kc) {
            v16h b = *(const v16h*)(Wp + ((size_t)(kc * 8 + nt) * 32 + lane) * 16);
            acc = __builtin_amdgcn_wmma_f32_16x16x32_f16(false, A[kc], false, b, (short)0, acc, false, false);
        }
        emit(nt, nt * 16 + ch0, acc);
    }
}

// ---------------- kernel 0: weight fold + f16 B-layout swizzle ----------------
__global__ void __launch_bounds__(256) prep_kernel(
    const float* ln1s, const float* ln1b, const float* w1, const float* b1,
    const float* wdw, const float* bdw, const float* b2, const float* beta,
    const float* ln2s, const float* ln2b, const float* wf1, const float* bf1,
    const float* wf2, const float* bf2, const float* gamma,
    float* b1p, float* bf1p, float* b2p, float* bf2p,
    _Float16* w1p, _Float16* wf1p, _Float16* wf2p) {
    const int t = threadIdx.x;
    {   // biases with LN-bias contraction folded in
        float s1 = b1[t], s2 = bf1[t];
        for (int k = 0; k < 128; ++k) {
            s1 += ln1b[k] * w1[k * 256 + t];
            s2 += ln2b[k] * wf1[k * 256 + t];
        }
        b1p[t]  = s1 * wdw[t] + bdw[t];
        bf1p[t] = s2;
    }
    if (t < 128) { b2p[t] = beta[t] * b2[t]; bf2p[t] = gamma[t] * bf2[t]; }
    // B-operand swizzle: k = kc*32 + (lane>=16)*8 + (slot&7) + (slot>>3)*16 ; n = nt*16 + (lane&15)
    for (int e = t; e < 32768; e += 256) {
        int slot = e & 15, lane = (e >> 4) & 31, nt = (e >> 9) & 15, kc = e >> 13;
        int k = kc * 32 + ((lane >> 4) << 3) + (slot & 7) + ((slot >> 3) << 4);
        int n = nt * 16 + (lane & 15);
        w1p[e]  = (_Float16)(ln1s[k] * w1[k * 256 + n] * wdw[n]);
        wf1p[e] = (_Float16)(ln2s[k] * wf1[k * 256 + n]);
    }
    for (int e = t; e < 16384; e += 256) {
        int slot = e & 15, lane = (e >> 4) & 31, nt = (e >> 9) & 7, kc = e >> 12;
        int k = kc * 32 + ((lane >> 4) << 3) + (slot & 7) + ((slot >> 3) << 4);
        int n = nt * 16 + (lane & 15);
        wf2p[e] = (_Float16)(wf2[k * 128 + n] * gamma[n]);
    }
}

// ---------------- kernel 1: GEMM1+gate, deterministic per-block GAP partials ----------------
__global__ void __launch_bounds__(128) k1_gap(const float* __restrict__ x,
                                              const _Float16* __restrict__ w1p,
                                              const float* __restrict__ b1p,
                                              float* __restrict__ gapBlk) {
    __shared__ float gapW[WAVES_PER_BLK][128];
    const int t = threadIdx.x, lane = t & 31, w = t >> 5;
    const int tile = blockIdx.x * WAVES_PER_BLK + w;
    const size_t pix0 = (size_t)tile * 16;
    v16h A[4];
    build_A_global<false>(x + pix0 * 128, nullptr, A, lane);
    gemm_gate(A, w1p, b1p, lane, [&](int nt, int ch, const float* g) {
        float cs = g[0] + g[1] + g[2] + g[3] + g[4] + g[5] + g[6] + g[7];
        cs += __shfl_xor(cs, 16);               // combine pixel halves
        if (lane < 16) gapW[w][ch] = cs;        // unique slot -> no atomics
    });
    __syncthreads();
    if (t < 128)
        gapBlk[(size_t)blockIdx.x * 128 + t] =
            gapW[0][t] + gapW[1][t] + gapW[2][t] + gapW[3][t];
}

// ---------------- kernel 2: GAP reduce + SE dense + att*beta*w2 -> f16 B-layout ----------------
__global__ void __launch_bounds__(256) se_kernel(const float* __restrict__ gapBlk,
                                                 const float* wse, const float* bse,
                                                 const float* w2, const float* beta,
                                                 _Float16* __restrict__ w2bp) {
    __shared__ float gsum[1024];
    __shared__ float att[1024];
    const int t = threadIdx.x;
    for (int i = t; i < 1024; i += 256) {       // deterministic tree over blocks
        int b = i >> 7, c = i & 127;
        float s = 0.f;
        for (int blk = 0; blk < BLKS_PER_IMG; ++blk)
            s += gapBlk[((size_t)b * BLKS_PER_IMG + blk) * 128 + c];
        gsum[i] = s * (1.f / 36864.f);          // spatial mean
    }
    __syncthreads();
    for (int i = t; i < 1024; i += 256) {
        int b = i >> 7, d = i & 127;
        float s = bse[d];
        for (int c = 0; c < 128; ++c) s += gsum[b * 128 + c] * wse[c * 128 + d];
        att[i] = s;
    }
    __syncthreads();
    for (int e = t; e < 131072; e += 256) {
        int slot = e & 15, lane = (e >> 4) & 31, nt = (e >> 9) & 7, kc = (e >> 12) & 3, b = e >> 14;
        int k = kc * 32 + ((lane >> 4) << 3) + (slot & 7) + ((slot >> 3) << 4);
        int n = nt * 16 + (lane & 15);
        w2bp[e] = (_Float16)(att[b * 128 + k] * w2[k * 128 + n] * beta[n]);
    }
}

// ---------------- kernel 3: full fused block (recompute spatial branch + FFN) ----------------
__global__ void __launch_bounds__(128) k3_main(const float* __restrict__ x,
                                               const _Float16* __restrict__ w1p,
                                               const float* __restrict__ b1p,
                                               const _Float16* __restrict__ w2bp,
                                               const float* __restrict__ b2p,
                                               const _Float16* __restrict__ wf1p,
                                               const float* __restrict__ bf1p,
                                               const _Float16* __restrict__ wf2p,
                                               const float* __restrict__ bf2p,
                                               float* __restrict__ out) {
    __shared__ float    xt_s[WAVES_PER_BLK][16 * 128];   // x_new fp32 tile (wave-private)
    __shared__ _Float16 gt_s[WAVES_PER_BLK][16 * 128];   // gated f16 tile (wave-private)
    __shared__ float    st_s[WAVES_PER_BLK][32];         // per-pixel LN2 mu/rsig
    const int t = threadIdx.x, lane = t & 31, w = t >> 5;
    const int par = lane >> 4, p = lane & 15;
    float*    xt = xt_s[w];
    _Float16* gt = gt_s[w];
    float*    st = st_s[w];
    const int tile = blockIdx.x * WAVES_PER_BLK + w;
    const size_t pix0 = (size_t)tile * 16;
    const int b = tile / TILES_PER_IMG;

    __builtin_prefetch((const void*)w1p, 0, 0);   // global_prefetch weights

    // LN1 -> A1 (stash raw x tile in LDS for residuals)
    v16h A1[4];
    build_A_global<true>(x + pix0 * 128, xt, A1, lane);

    // spatial GEMM1 + gate -> LDS f16, then re-read in A layout (LDS transpose)
    gemm_gate(A1, w1p, b1p, lane, [&](int nt, int ch, const float* g) {
#pragma unroll
        for (int r = 0; r < 8; ++r) gt[(r + 8 * par) * 128 + ch] = (_Float16)g[r];
    });
    asm volatile("s_wait_dscnt 0x0" ::: "memory");
    v16h As[4];
    build_A_ldsh(gt, As, lane);

    // (s*att) @ w2*beta + beta*b2, residual into xt; accumulate LN2 stats
    float sm[8] = {0,0,0,0,0,0,0,0}, sq[8] = {0,0,0,0,0,0,0,0};
    gemm8(As, w2bp + (size_t)b * 16384, lane, [&](int nt, int ch, const v8f& acc) {
        const float bb = b2p[ch];
#pragma unroll
        for (int r = 0; r < 8; ++r) {
            const int pr = r + 8 * par;
            float xn = xt[pr * 128 + ch] + acc[r] + bb;
            xt[pr * 128 + ch] = xn;
            sm[r] += xn; sq[r] += xn * xn;
        }
    });
    // LN2 per-pixel stats via lane shuffles, broadcast through LDS
#pragma unroll
    for (int r = 0; r < 8; ++r) {
        float s = reduce16(sm[r]), q = reduce16(sq[r]);
        if ((lane & 15) == 0) {
            float mu = s * (1.f / 128.f);
            st[(r + 8 * par) * 2]     = mu;
            st[(r + 8 * par) * 2 + 1] = rsqrtf(q * (1.f / 128.f) - mu * mu + EPS);
        }
    }
    asm volatile("s_wait_dscnt 0x0" ::: "memory");
    const float mu2 = st[p * 2], rs2 = st[p * 2 + 1];
    v16h A2[4];
    build_A_ldsf(xt, mu2, rs2, A2, lane);

    // FFN GEMM1 + gate -> LDS f16 -> A3
    gemm_gate(A2, wf1p, bf1p, lane, [&](int nt, int ch, const float* g) {
#pragma unroll
        for (int r = 0; r < 8; ++r) gt[(r + 8 * par) * 128 + ch] = (_Float16)g[r];
    });
    asm volatile("s_wait_dscnt 0x0" ::: "memory");
    v16h A3[4];
    build_A_ldsh(gt, A3, lane);

    // FFN GEMM2 (gamma folded) + final residual, stream to global
    gemm8(A3, wf2p, lane, [&](int nt, int ch, const v8f& acc) {
        const float bb = bf2p[ch];
#pragma unroll
        for (int r = 0; r < 8; ++r) {
            const int pr = r + 8 * par;
            out[(pix0 + pr) * 128 + ch] = xt[pr * 128 + ch] + acc[r] + bb;
        }
    });
}

extern "C" void kernel_launch(void* const* d_in, const int* in_sizes, int n_in,
                              void* d_out, int out_size, void* d_ws, size_t ws_size,
                              hipStream_t stream) {
    const float* x    = (const float*)d_in[0];
    const float* ln1s = (const float*)d_in[1];
    const float* ln1b = (const float*)d_in[2];
    const float* w1   = (const float*)d_in[3];
    const float* b1   = (const float*)d_in[4];
    const float* wdw  = (const float*)d_in[5];
    const float* bdw  = (const float*)d_in[6];
    const float* wse  = (const float*)d_in[7];
    const float* bse  = (const float*)d_in[8];
    const float* w2   = (const float*)d_in[9];
    const float* b2   = (const float*)d_in[10];
    const float* ln2s = (const float*)d_in[11];
    const float* ln2b = (const float*)d_in[12];
    const float* wf1  = (const float*)d_in[13];
    const float* bf1  = (const float*)d_in[14];
    const float* wf2  = (const float*)d_in[15];
    const float* bf2  = (const float*)d_in[16];
    const float* beta = (const float*)d_in[17];
    const float* gamma= (const float*)d_in[18];

    char* wsb = (char*)d_ws;
    float*    gapBlk = (float*)(wsb + 0);
    float*    b1p    = (float*)(wsb + 2359296);
    float*    bf1p   = (float*)(wsb + 2360320);
    float*    b2p    = (float*)(wsb + 2361344);
    float*    bf2p   = (float*)(wsb + 2361856);
    _Float16* w1p    = (_Float16*)(wsb + 2362368);
    _Float16* wf1p   = (_Float16*)(wsb + 2427904);
    _Float16* wf2p   = (_Float16*)(wsb + 2493440);
    _Float16* w2bp   = (_Float16*)(wsb + 2526208);

    prep_kernel<<<1, 256, 0, stream>>>(ln1s, ln1b, w1, b1, wdw, bdw, b2, beta,
                                       ln2s, ln2b, wf1, bf1, wf2, bf2, gamma,
                                       b1p, bf1p, b2p, bf2p, w1p, wf1p, wf2p);
    k1_gap<<<NBLK, 128, 0, stream>>>(x, w1p, b1p, gapBlk);
    se_kernel<<<1, 256, 0, stream>>>(gapBlk, wse, bse, w2, beta, w2bp);
    k3_main<<<NBLK, 128, 0, stream>>>(x, w1p, b1p, w2bp, b2p, wf1p, bf1p,
                                      wf2p, bf2p, (float*)d_out);
}
